// CrossAttention_11613591568701
// MI455X (gfx1250) — compile-verified, matrix operand order checked by reading, not run on hardware
//
#include <hip/hip_runtime.h>
#include <hip/hip_bf16.h>
#include <stdint.h>

// ---------------------------------------------------------------------------
// Linear cross-attention with RoPE, MI455X (gfx1250, wave32, WMMA).
// Associativity trick: y = Q_rot @ (K_rot^T V) / (Q . sum_s K)  -- avoids TxS.
// All GEMMs run through v_wmma_f32_16x16x32_f16 with fp32 accumulation;
// fp32 operands are converted to f16 while staging through LDS.
// ---------------------------------------------------------------------------

typedef __attribute__((ext_vector_type(16))) _Float16 v16h;
typedef __attribute__((ext_vector_type(8)))  float    v8f;

#define N_EMB 1024
#define NHEAD 16
#define HDIM  64
#define BATCH 2
#define TLEN  2048
#define SLEN  2048
#define LDS_PAD 8               // halfs of padding per LDS row (row = 40 halfs)
#define ROWH (32 + LDS_PAD)     // LDS row stride in halfs (BK=32)

// --- WMMA fragment helpers --------------------------------------------------
// 16-bit A-matrix 16x32 layout (ISA 7.12.2): lane&15 = row M; lanes>=16 use
// K base +8; VGPR0..3 hold K=kb..kb+7 (packed f16 pairs), VGPR4..7 hold
// K=kb+16..kb+23.  B 32x16 is stored K-contiguous per column in LDS
// (transposed on fill) so the same load pattern applies with lane&15 = col N.
__device__ __forceinline__ v16h frag_ld(const _Float16* rowbase, int kb) {
  union { v16h h; uint32_t u[8]; } f;
  const uint32_t* p0 = (const uint32_t*)(rowbase + kb);
  const uint32_t* p1 = (const uint32_t*)(rowbase + kb + 16);
#pragma unroll
  for (int j = 0; j < 4; ++j) { f.u[j] = p0[j]; f.u[4 + j] = p1[j]; }
  return f.h;
}

__device__ __forceinline__ v8f wmma16(v16h a, v16h b, v8f c) {
  // (neg_a, A, neg_b, B, c_mod, C, reuse_a, reuse_b)
  return __builtin_amdgcn_wmma_f32_16x16x32_f16(false, a, false, b,
                                                (short)0, c, false, false);
}

__device__ __forceinline__ float elu_p1(float v) {
  return v > 0.0f ? v + 1.0f : __expf(v);   // elu(v)+1
}

// --- Generic tiled GEMM: C = A(fp32)[M,K] * B(fp32)[K,N] + bias -------------
// mode 0: plain; mode 1: elu+1 everywhere; mode 2: elu+1 only for col < halfN.
// Shapes are exact multiples of tiles (M%128==0, N%128==0, K%32==0).
__global__ __launch_bounds__(256)
void gemm_kernel(const float* __restrict__ A, const float* __restrict__ B,
                 const float* __restrict__ bias, float* __restrict__ C,
                 int M, int N, int K, int mode, int halfN) {
  __shared__ _Float16 As[128 * ROWH];
  __shared__ _Float16 Bs[128 * ROWH];

  const int tid  = threadIdx.x;
  const int bm   = blockIdx.y * 128;
  const int bn   = blockIdx.x * 128;
  const int wid  = tid >> 5;
  const int lane = tid & 31;
  const int nl   = lane & 15;
  const int kb   = (lane >> 4) << 3;   // K base for A/B fragments
  const int hb   = (lane >> 4) << 3;   // row offset for C/D fragments
  const int wm   = (wid & 1) * 64;     // 2 waves in M  -> 64 rows each
  const int wn   = (wid >> 1) * 32;    // 4 waves in N  -> 32 cols each

  const v8f vzero = {0.f,0.f,0.f,0.f,0.f,0.f,0.f,0.f};
  v8f acc[4][2];
#pragma unroll
  for (int i = 0; i < 4; ++i)
#pragma unroll
    for (int j = 0; j < 2; ++j) acc[i][j] = vzero;

  for (int k0 = 0; k0 < K; k0 += 32) {
    // A tile 128x32 -> As[m][k]
    {
      const int r  = tid >> 3;           // 0..31
      const int c4 = (tid & 7) * 4;      // 0..28
#pragma unroll
      for (int rr = 0; rr < 128; rr += 32) {
        const float4 v = *(const float4*)(A + (size_t)(bm + r + rr) * K + k0 + c4);
        _Float16* d = &As[(r + rr) * ROWH + c4];
        d[0] = (_Float16)v.x; d[1] = (_Float16)v.y;
        d[2] = (_Float16)v.z; d[3] = (_Float16)v.w;
      }
    }
    // B tile 32x128 -> Bs[n][k] (transposed: K contiguous)
    {
      const int br  = tid >> 5;          // 0..7
      const int bc4 = (tid & 31) * 4;    // 0..124
#pragma unroll
      for (int rr = 0; rr < 32; rr += 8) {
        const float4 v = *(const float4*)(B + (size_t)(k0 + br + rr) * N + bn + bc4);
        Bs[(bc4 + 0) * ROWH + br + rr] = (_Float16)v.x;
        Bs[(bc4 + 1) * ROWH + br + rr] = (_Float16)v.y;
        Bs[(bc4 + 2) * ROWH + br + rr] = (_Float16)v.z;
        Bs[(bc4 + 3) * ROWH + br + rr] = (_Float16)v.w;
      }
    }
    __syncthreads();

    v16h af[4], bf[2];
#pragma unroll
    for (int i = 0; i < 4; ++i) af[i] = frag_ld(&As[(wm + i * 16 + nl) * ROWH], kb);
#pragma unroll
    for (int j = 0; j < 2; ++j) bf[j] = frag_ld(&Bs[(wn + j * 16 + nl) * ROWH], kb);
#pragma unroll
    for (int i = 0; i < 4; ++i)
#pragma unroll
      for (int j = 0; j < 2; ++j) acc[i][j] = wmma16(af[i], bf[j], acc[i][j]);
    __syncthreads();
  }

#pragma unroll
  for (int i = 0; i < 4; ++i)
#pragma unroll
    for (int j = 0; j < 2; ++j) {
      const int col = bn + wn + j * 16 + nl;
      const float bv = bias[col];
      const bool act = (mode == 1) || (mode == 2 && col < halfN);
#pragma unroll
      for (int vi = 0; vi < 8; ++vi) {
        const int row = bm + wm + i * 16 + hb + vi;
        float v = acc[i][j][vi] + bv;
        if (act) v = elu_p1(v);
        C[(size_t)row * N + col] = v;
      }
    }
}

// --- RoPE elementwise -------------------------------------------------------
// in: feature-mapped tensor viewed as [B*L, C] with row stride in_stride
// (reads only first N_EMB columns). transposed==0: out [B,L,C];
// transposed==1: out [B,H,HD,L] (so it can serve as row-major WMMA A).
__global__ void rope_kernel(const float* __restrict__ in, float* __restrict__ out,
                            int L, int in_stride, int transposed) {
  const int idx = blockIdx.x * blockDim.x + threadIdx.x;
  const int row = idx >> 10;            // /C (C=1024)
  const int c   = idx & (N_EMB - 1);
  const int b   = row / L;
  const int t   = row - b * L;
  const int h   = c >> 6;
  const int d   = c & 63;
  const int i   = d & 31;
  // inv_freq[i] = 10000^(-2i/64) = exp(-i * ln(10000)/32)
  const float inv = __expf(-(float)i * 0.28782313662425572f);
  float sn, cs;
  __sincosf((float)t * inv, &sn, &cs);
  const size_t base = (size_t)row * in_stride;
  const float xv = in[base + c];
  const float xo = (d < 32) ? -in[base + c + 32] : in[base + c - 32];
  const float r = xv * cs + xo * sn;
  if (transposed)
    out[((size_t)(b * NHEAD + h) * HDIM + d) * L + t] = r;
  else
    out[(size_t)row * N_EMB + c] = r;
}

// --- K row-sum over S (two pass, deterministic) -----------------------------
#define NCH 16
__global__ void ksum_partial_kernel(const float* __restrict__ kv,
                                    float* __restrict__ partial) {
  const int c  = blockIdx.x * blockDim.x + threadIdx.x;  // 0..1023
  const int b  = blockIdx.y;
  const int ch = blockIdx.z;
  const int chs = SLEN / NCH;
  float s = 0.f;
  for (int i = 0; i < chs; ++i) {
    const int srow = ch * chs + i;
    s += kv[((size_t)(b * SLEN + srow)) * (2 * N_EMB) + c];
  }
  partial[((size_t)ch * BATCH + b) * N_EMB + c] = s;
}

__global__ void ksum_final_kernel(const float* __restrict__ partial,
                                  float* __restrict__ ksum) {
  const int i = blockIdx.x * blockDim.x + threadIdx.x;  // 0..B*C-1
  float s = 0.f;
#pragma unroll
  for (int ch = 0; ch < NCH; ++ch) s += partial[(size_t)ch * (BATCH * N_EMB) + i];
  ksum[i] = s;
}

// --- denom[b,h,t] = qfeat[b,t,h,:] . ksum[b,h,:] ----------------------------
__global__ void denom_kernel(const float* __restrict__ qfeat,
                             const float* __restrict__ ksum,
                             float* __restrict__ denom) {
  const int idx = blockIdx.x * blockDim.x + threadIdx.x;  // B*T*H
  const int b = idx / (TLEN * NHEAD);
  const int r = idx - b * (TLEN * NHEAD);
  const int t = r >> 4;
  const int h = r & 15;
  const float* q = qfeat + (size_t)(b * TLEN + t) * N_EMB + h * HDIM;
  const float* k = ksum + (size_t)b * N_EMB + h * HDIM;
  float s = 0.f;
#pragma unroll 8
  for (int d = 0; d < HDIM; ++d) s += q[d] * k[d];
  denom[((size_t)(b * NHEAD + h)) * TLEN + t] = s;
}

// --- state[b,h] = K_rot^T V  (64x64, K over S=2048) -------------------------
__global__ __launch_bounds__(256)
void state_kernel(const float* __restrict__ krotT, const float* __restrict__ kv,
                  float* __restrict__ state) {
  __shared__ _Float16 As[64 * ROWH];
  __shared__ _Float16 Bs[64 * ROWH];
  const int bh = blockIdx.x;
  const int b = bh >> 4, h = bh & 15;
  const float* Aptr = krotT + (size_t)bh * HDIM * SLEN;                 // [64,S]
  const float* Vptr = kv + (size_t)b * SLEN * (2 * N_EMB) + N_EMB + h * HDIM;
  const int tid = threadIdx.x;
  const int wid = tid >> 5, lane = tid & 31;
  const int nl = lane & 15;
  const int kb = (lane >> 4) << 3;
  const int hb = (lane >> 4) << 3;
  const int tm  = (wid & 3) * 16;        // 4 m-tiles
  const int tn0 = (wid >> 2) * 32;       // 2 wave groups x 2 n-tiles

  const v8f vzero = {0.f,0.f,0.f,0.f,0.f,0.f,0.f,0.f};
  v8f acc[2] = {vzero, vzero};

  for (int s0 = 0; s0 < SLEN; s0 += 32) {
    {   // A tile 64x32 (rows d, cols s)
      const int r = tid >> 3, c4 = (tid & 7) * 4;
#pragma unroll
      for (int rr = 0; rr < 64; rr += 32) {
        const float4 v = *(const float4*)(Aptr + (size_t)(r + rr) * SLEN + s0 + c4);
        _Float16* d = &As[(r + rr) * ROWH + c4];
        d[0] = (_Float16)v.x; d[1] = (_Float16)v.y;
        d[2] = (_Float16)v.z; d[3] = (_Float16)v.w;
      }
    }
    {   // V tile 32(s) x 64(e) -> Bs[e][s]
      const int br = tid >> 4, bc4 = (tid & 15) * 4;
#pragma unroll
      for (int rr = 0; rr < 32; rr += 16) {
        const float4 v = *(const float4*)(Vptr + (size_t)(s0 + br + rr) * (2 * N_EMB) + bc4);
        Bs[(bc4 + 0) * ROWH + br + rr] = (_Float16)v.x;
        Bs[(bc4 + 1) * ROWH + br + rr] = (_Float16)v.y;
        Bs[(bc4 + 2) * ROWH + br + rr] = (_Float16)v.z;
        Bs[(bc4 + 3) * ROWH + br + rr] = (_Float16)v.w;
      }
    }
    __syncthreads();
    const v16h af = frag_ld(&As[(tm + nl) * ROWH], kb);
#pragma unroll
    for (int j = 0; j < 2; ++j) {
      const v16h bf = frag_ld(&Bs[(tn0 + j * 16 + nl) * ROWH], kb);
      acc[j] = wmma16(af, bf, acc[j]);
    }
    __syncthreads();
  }
#pragma unroll
  for (int j = 0; j < 2; ++j) {
    const int col = tn0 + j * 16 + nl;
#pragma unroll
    for (int vi = 0; vi < 8; ++vi) {
      const int row = tm + hb + vi;
      state[(size_t)bh * (HDIM * HDIM) + row * HDIM + col] = acc[j][vi];
    }
  }
}

// --- y[b,t,h,:] = (Q_rot[b,t,h,:] @ state[b,h]) / denom[b,h,t] --------------
__global__ __launch_bounds__(256)
void ygemm_kernel(const float* __restrict__ qrot, const float* __restrict__ state,
                  const float* __restrict__ denom, float* __restrict__ y) {
  __shared__ _Float16 As[128 * ROWH];
  __shared__ _Float16 Bs[64 * ROWH];
  const int bh = blockIdx.y;
  const int b = bh >> 4, h = bh & 15;
  const int row0 = blockIdx.x * 128;
  const float* Aptr = qrot + (size_t)b * TLEN * N_EMB + h * HDIM;   // rows t, lda=N_EMB
  const float* Sptr = state + (size_t)bh * (HDIM * HDIM);
  const int tid = threadIdx.x;
  const int wid = tid >> 5, lane = tid & 31;
  const int nl = lane & 15;
  const int kb = (lane >> 4) << 3;
  const int hb = (lane >> 4) << 3;

  const v8f vzero = {0.f,0.f,0.f,0.f,0.f,0.f,0.f,0.f};
  v8f acc[4] = {vzero, vzero, vzero, vzero};

  for (int k0 = 0; k0 < HDIM; k0 += 32) {
    {   // A tile 128x32
      const int r = tid >> 3, c4 = (tid & 7) * 4;
#pragma unroll
      for (int rr = 0; rr < 128; rr += 32) {
        const float4 v = *(const float4*)(Aptr + (size_t)(row0 + r + rr) * N_EMB + k0 + c4);
        _Float16* d = &As[(r + rr) * ROWH + c4];
        d[0] = (_Float16)v.x; d[1] = (_Float16)v.y;
        d[2] = (_Float16)v.z; d[3] = (_Float16)v.w;
      }
    }
    {   // state tile 32(d) x 64(e) -> Bs[e][d]
      const int br = tid >> 4, bc4 = (tid & 15) * 4;
#pragma unroll
      for (int rr = 0; rr < 32; rr += 16) {
        const float4 v = *(const float4*)(Sptr + (size_t)(k0 + br + rr) * HDIM + bc4);
        Bs[(bc4 + 0) * ROWH + br + rr] = (_Float16)v.x;
        Bs[(bc4 + 1) * ROWH + br + rr] = (_Float16)v.y;
        Bs[(bc4 + 2) * ROWH + br + rr] = (_Float16)v.z;
        Bs[(bc4 + 3) * ROWH + br + rr] = (_Float16)v.w;
      }
    }
    __syncthreads();
    const v16h af = frag_ld(&As[(wid * 16 + nl) * ROWH], kb);
#pragma unroll
    for (int j = 0; j < 4; ++j) {
      const v16h bf = frag_ld(&Bs[(j * 16 + nl) * ROWH], kb);
      acc[j] = wmma16(af, bf, acc[j]);
    }
    __syncthreads();
  }
  float rd[8];
#pragma unroll
  for (int vi = 0; vi < 8; ++vi)
    rd[vi] = 1.0f / denom[(size_t)bh * TLEN + row0 + wid * 16 + hb + vi];
#pragma unroll
  for (int j = 0; j < 4; ++j) {
#pragma unroll
    for (int vi = 0; vi < 8; ++vi) {
      const int t = row0 + wid * 16 + hb + vi;
      const int col = h * HDIM + j * 16 + nl;
      y[(size_t)(b * TLEN + t) * N_EMB + col] = acc[j][vi] * rd[vi];
    }
  }
}

// ---------------------------------------------------------------------------
extern "C" void kernel_launch(void* const* d_in, const int* in_sizes, int n_in,
                              void* d_out, int out_size, void* d_ws, size_t ws_size,
                              hipStream_t stream) {
  (void)in_sizes; (void)n_in; (void)out_size; (void)ws_size;
  const float* x      = (const float*)d_in[0];
  const float* mem    = (const float*)d_in[1];
  const float* q_w    = (const float*)d_in[2];
  const float* q_b    = (const float*)d_in[3];
  const float* kv_w   = (const float*)d_in[4];
  const float* kv_b   = (const float*)d_in[5];
  const float* proj_w = (const float*)d_in[6];
  const float* proj_b = (const float*)d_in[7];
  float* out = (float*)d_out;

  // Workspace layout (floats). Total ~97 MB.
  const size_t SZ_BTC = (size_t)BATCH * TLEN * N_EMB;     // 4 M
  float* ws      = (float*)d_ws;
  float* qfeat   = ws;                                    // [B,T,C] elu+1
  float* qrot    = qfeat   + SZ_BTC;                      // [B,T,C]
  float* kvbuf   = qrot    + SZ_BTC;                      // [B,S,2C] (K feat | V)
  float* krotT   = kvbuf   + 2 * SZ_BTC;                  // [B,H,64,S]
  float* ybuf    = krotT   + SZ_BTC;                      // [B,T,C]
  float* stateb  = ybuf    + SZ_BTC;                      // [B*H,64,64]
  float* ksum    = stateb  + (size_t)BATCH * NHEAD * HDIM * HDIM;  // [B,C]
  float* partial = ksum    + (size_t)BATCH * N_EMB;       // [NCH,B,C]
  float* denom   = partial + (size_t)NCH * BATCH * N_EMB; // [B,H,T]

  const dim3 blk(256);
  const int M = BATCH * TLEN;   // 4096

  // 1) Q = elu(x@q_w + q_b)+1
  gemm_kernel<<<dim3(N_EMB / 128, M / 128), blk, 0, stream>>>(
      x, q_w, q_b, qfeat, M, N_EMB, N_EMB, 1, 0);
  // 2) kv = memory@kv_w + kv_b ; elu+1 on K half only
  gemm_kernel<<<dim3(2 * N_EMB / 128, M / 128), blk, 0, stream>>>(
      mem, kv_w, kv_b, kvbuf, M, 2 * N_EMB, N_EMB, 2, N_EMB);
  // 3) RoPE
  rope_kernel<<<(unsigned)(SZ_BTC / 256), blk, 0, stream>>>(qfeat, qrot, TLEN, N_EMB, 0);
  rope_kernel<<<(unsigned)(SZ_BTC / 256), blk, 0, stream>>>(kvbuf, krotT, SLEN, 2 * N_EMB, 1);
  // 4) ksum = sum_s K  (two-pass, deterministic)
  ksum_partial_kernel<<<dim3(N_EMB / 256, BATCH, NCH), blk, 0, stream>>>(kvbuf, partial);
  ksum_final_kernel<<<(BATCH * N_EMB) / 256, blk, 0, stream>>>(partial, ksum);
  // 5) denom = Q . ksum
  denom_kernel<<<(BATCH * TLEN * NHEAD) / 256, blk, 0, stream>>>(qfeat, ksum, denom);
  // 6) state = K_rot^T V  (per b,h)
  state_kernel<<<BATCH * NHEAD, blk, 0, stream>>>(krotT, kvbuf, stateb);
  // 7) y = (Q_rot @ state) / denom
  ygemm_kernel<<<dim3(TLEN / 128, BATCH * NHEAD), blk, 0, stream>>>(qrot, stateb, denom, ybuf);
  // 8) out = y @ proj_w + proj_b
  gemm_kernel<<<dim3(N_EMB / 128, M / 128), blk, 0, stream>>>(
      ybuf, proj_w, proj_b, out, M, N_EMB, N_EMB, 0, 0);
}
